// MAMBA_block_MY_38001870635999
// MI455X (gfx1250) — matrix operand order repaired and checked by reading, not compiled, not run on hardware
//
#include <hip/hip_runtime.h>
#include <hip/hip_bf16.h>

#define SEQ     4096
#define DMODEL  512
#define DINNER  1024
#define DSTATE  16
#define DTRANK  32
#define ATTD    128
#define NCHUNK  64
#define CLEN    64   // NCHUNK * CLEN == SEQ

typedef __attribute__((ext_vector_type(16))) __bf16          v16bf;
typedef __attribute__((ext_vector_type(16))) unsigned short  v16us;
typedef __attribute__((ext_vector_type(8)))  float           v8f;

union Frag { unsigned short us[16]; v16us u; v16bf b; };

__device__ __forceinline__ unsigned short f32_to_bf16_rne(float f) {
  unsigned int u = __float_as_uint(f);
  u += 0x7FFFu + ((u >> 16) & 1u);           // round-to-nearest-even
  return (unsigned short)(u >> 16);
}
__device__ __forceinline__ float silu_f(float x) { return x / (1.f + __expf(-x)); }

// ---------------------------------------------------------------------------
// Fragment packing: convert f32 -> bf16 ONCE per element, emitting data in
// WMMA fragment order so each lane's 16 elements are 32 contiguous bytes.
//
// A fragment (16x32 tile): lanes 0-15 -> M=lane, K={0..7,16..23};
//                          lanes 16-31 -> M=lane-16, K={8..15,24..31}
// Layout: P[(mTile*kSteps + ks)*32 + lane][16]
// ---------------------------------------------------------------------------
__global__ __launch_bounds__(256) void pack_a_kernel(
    const float* __restrict__ A, int lda, int kSteps,
    unsigned short* __restrict__ P)
{
  int idx  = blockIdx.x * 256 + threadIdx.x;   // (mTile*kSteps + ks)*32 + lane
  int lane = idx & 31;
  int tk   = idx >> 5;
  int ks   = tk % kSteps;
  int mt   = tk / kSteps;
  int half = lane >> 4;
  int m    = lane & 15;
  const float* src = A + (size_t)(mt * 16 + m) * lda + ks * 32;
  Frag f;
  #pragma unroll
  for (int e = 0; e < 8; ++e) f.us[e]     = f32_to_bf16_rne(src[half * 8 + e]);
  #pragma unroll
  for (int e = 0; e < 8; ++e) f.us[8 + e] = f32_to_bf16_rne(src[16 + half * 8 + e]);
  *((v16us*)(P + (size_t)idx * 16)) = f.u;
}

// B fragment (32x16 tile of W^T): lanes 0-15 -> N=lane, K={0..7,16..23};
//                                 lanes 16-31 -> N=lane-16, K={8..15,24..31}
// Layout: P[(nTile16*kSteps + ks)*32 + lane][16]   (same K order as A)
__global__ __launch_bounds__(256) void pack_w_kernel(
    const float* __restrict__ W, int ldw, int kSteps,
    unsigned short* __restrict__ P)
{
  int idx  = blockIdx.x * 256 + threadIdx.x;   // (nTile*kSteps + ks)*32 + lane
  int lane = idx & 31;
  int tk   = idx >> 5;
  int ks   = tk % kSteps;
  int nt   = tk / kSteps;
  int half = lane >> 4;
  int n    = lane & 15;
  const float* src = W + (size_t)(nt * 16 + n) * ldw + ks * 32;
  Frag f;
  #pragma unroll
  for (int e = 0; e < 8; ++e) f.us[e]     = f32_to_bf16_rne(src[half * 8 + e]);
  #pragma unroll
  for (int e = 0; e < 8; ++e) f.us[8 + e] = f32_to_bf16_rne(src[16 + half * 8 + e]);
  *((v16us*)(P + (size_t)idx * 16)) = f.u;
}

// ---------------------------------------------------------------------------
// GEMM on pre-packed fragments: Out[M,N] = act(A @ W^T + bias)
// One wave -> 16x64 strip, register-double-buffered software pipeline:
// fragments for step ks+1 are in flight while the 4 WMMAs of step ks issue.
// __launch_bounds__(256,1) lifts the 128-VGPR occupancy cap so the pipeline
// stays in registers (no LDS spill); ~140 VGPRs -> still ~7 waves/SIMD.
// ---------------------------------------------------------------------------
__global__ __launch_bounds__(256, 1) void gemm_frag_wmma_kernel(
    const unsigned short* __restrict__ Apk,
    const unsigned short* __restrict__ Wpk,
    const float* __restrict__ bias,
    float* __restrict__ Out, int ldo, int kSteps, int act)
{
  const int lane = threadIdx.x & 31;
  const int wave = threadIdx.x >> 5;
  const int mt   = blockIdx.y * 8 + wave;      // 16-row tile index
  const int nt0  = blockIdx.x * 4;             // first 16-col tile index

  const v16us* ap = (const v16us*)Apk + (size_t)mt * kSteps * 32 + lane;
  const v16us* bp[4];
  #pragma unroll
  for (int nb = 0; nb < 4; ++nb)
    bp[nb] = (const v16us*)Wpk + (size_t)(nt0 + nb) * kSteps * 32 + lane;

  v8f acc[4] = {};
  Frag af[2];
  Frag bf[2][4];

  // prologue: stage K-step 0
  af[0].u = ap[0];
  #pragma unroll
  for (int nb = 0; nb < 4; ++nb) bf[0][nb].u = bp[nb][0];

  int ks = 0;
  for (; ks + 1 < kSteps; ++ks) {
    const int cur = ks & 1;
    const int nxt = cur ^ 1;
    // issue next-step fragment loads before this step's math
    af[nxt].u = ap[(size_t)(ks + 1) * 32];
    #pragma unroll
    for (int nb = 0; nb < 4; ++nb) bf[nxt][nb].u = bp[nb][(size_t)(ks + 1) * 32];
    // speculative global_prefetch_b8 two steps ahead (stays inside workspace)
    __builtin_prefetch(ap + (size_t)(ks + 2) * 32, 0, 1);
    __builtin_prefetch(bp[0] + (size_t)(ks + 2) * 32, 0, 1);
    #pragma unroll
    for (int nb = 0; nb < 4; ++nb) {
      acc[nb] = __builtin_amdgcn_wmma_f32_16x16x32_bf16(
          false, af[cur].b, false, bf[cur][nb].b, (short)0, acc[nb], false, false);
    }
  }
  {   // peeled final K-step (no loads, no guards)
    const int cur = ks & 1;
    #pragma unroll
    for (int nb = 0; nb < 4; ++nb) {
      acc[nb] = __builtin_amdgcn_wmma_f32_16x16x32_bf16(
          false, af[cur].b, false, bf[cur][nb].b, (short)0, acc[nb], false, false);
    }
  }

  // C/D layout: VGPR r -> row r (lanes 0-15) / row 8+r (lanes 16-31), col = lane&15
  const int col   = lane & 15;
  const int rbase = (lane < 16) ? 0 : 8;
  #pragma unroll
  for (int nb = 0; nb < 4; ++nb) {
    #pragma unroll
    for (int r = 0; r < 8; ++r) {
      int row = mt * 16 + rbase + r;
      int c   = (nt0 + nb) * 16 + col;
      float v = acc[nb][r];
      if (bias) v += bias[c];
      if (act == 1)      v = silu_f(v);
      else if (act == 2) v = tanhf(v);
      else if (act == 3) v = (v > 20.f) ? v : __logf(1.f + __expf(v)); // softplus
      Out[(size_t)row * ldo + c] = v;
    }
  }
}

// ---------------------------------------------------------------------------
// Depthwise causal conv (kernel 4, left pad 3) + SiLU.  xin_raw = xz[:, 0:1024].
// ---------------------------------------------------------------------------
__global__ __launch_bounds__(256) void conv_silu_kernel(
    const float* __restrict__ xz, const float* __restrict__ cw,
    const float* __restrict__ cb, float* __restrict__ xin)
{
  int idx = blockIdx.x * 256 + threadIdx.x;     // over SEQ*DINNER
  if (idx >= SEQ * DINNER) return;
  int d = idx & (DINNER - 1);
  int t = idx >> 10;
  float acc = cb[d];
  #pragma unroll
  for (int j = 0; j < 4; ++j) {
    int tt = t - 3 + j;
    if (tt >= 0) acc += cw[d * 4 + j] * xz[(size_t)tt * (2 * DINNER) + d];
  }
  xin[idx] = silu_f(acc);
}

// ---------------------------------------------------------------------------
// Selective scan, chunked (associative in (a,b)).  dA/dBu recomputed on the fly.
// thread = channel d (16 states in registers); blockIdx.y = chunk.
// ---------------------------------------------------------------------------
__global__ __launch_bounds__(256) void scan_local_kernel(
    const float* __restrict__ dt, const float* __restrict__ xdbl,
    const float* __restrict__ xin, const float* __restrict__ A_log,
    float* __restrict__ prodA, float* __restrict__ hend)
{
  int d = blockIdx.x * 256 + threadIdx.x;
  int c = blockIdx.y;
  float Av[DSTATE];
  #pragma unroll
  for (int s = 0; s < DSTATE; ++s) Av[s] = -__expf(A_log[d * DSTATE + s]);
  float h[DSTATE] = {}, p[DSTATE];
  #pragma unroll
  for (int s = 0; s < DSTATE; ++s) p[s] = 1.f;
  for (int i = 0; i < CLEN; ++i) {
    int t = c * CLEN + i;
    float dtv = dt[(size_t)t * DINNER + d];
    float xv  = xin[(size_t)t * DINNER + d];
    const float* xr = xdbl + (size_t)t * (DTRANK + 2 * DSTATE);
    #pragma unroll
    for (int s = 0; s < DSTATE; ++s) {
      float a = __expf(dtv * Av[s]);
      float b = dtv * xr[DTRANK + s] * xv;
      h[s] = a * h[s] + b;
      p[s] *= a;
    }
  }
  size_t base = ((size_t)c * DINNER + d) * DSTATE;
  #pragma unroll
  for (int s = 0; s < DSTATE; ++s) { prodA[base + s] = p[s]; hend[base + s] = h[s]; }
}

__global__ __launch_bounds__(256) void scan_carry_kernel(
    const float* __restrict__ prodA, const float* __restrict__ hend,
    float* __restrict__ carry)
{
  int d = blockIdx.x * 256 + threadIdx.x;
  float hc[DSTATE] = {};
  for (int c = 0; c < NCHUNK; ++c) {
    size_t base = ((size_t)c * DINNER + d) * DSTATE;
    #pragma unroll
    for (int s = 0; s < DSTATE; ++s) {
      carry[base + s] = hc[s];
      hc[s] = prodA[base + s] * hc[s] + hend[base + s];
    }
  }
}

__global__ __launch_bounds__(256) void scan_apply_kernel(
    const float* __restrict__ dt, const float* __restrict__ xdbl,
    const float* __restrict__ xin, const float* __restrict__ A_log,
    const float* __restrict__ carry, const float* __restrict__ Dp,
    const float* __restrict__ xz, float* __restrict__ y)
{
  int d = blockIdx.x * 256 + threadIdx.x;
  int c = blockIdx.y;
  float Av[DSTATE];
  #pragma unroll
  for (int s = 0; s < DSTATE; ++s) Av[s] = -__expf(A_log[d * DSTATE + s]);
  float h[DSTATE];
  size_t base = ((size_t)c * DINNER + d) * DSTATE;
  #pragma unroll
  for (int s = 0; s < DSTATE; ++s) h[s] = carry[base + s];
  float Dv = Dp[d];
  for (int i = 0; i < CLEN; ++i) {
    int t = c * CLEN + i;
    float dtv = dt[(size_t)t * DINNER + d];
    float xv  = xin[(size_t)t * DINNER + d];
    const float* xr = xdbl + (size_t)t * (DTRANK + 2 * DSTATE);
    float acc = 0.f;
    #pragma unroll
    for (int s = 0; s < DSTATE; ++s) {
      float a = __expf(dtv * Av[s]);
      h[s] = a * h[s] + dtv * xr[DTRANK + s] * xv;
      acc += h[s] * xr[DTRANK + DSTATE + s];
    }
    float yv = acc + xv * Dv;
    float zv = xz[(size_t)t * (2 * DINNER) + DINNER + d];
    y[(size_t)t * DINNER + d] = yv * silu_f(zv);   // fuse gating for out-GEMM
  }
}

// ---------------------------------------------------------------------------
// Attention pooling tail
// ---------------------------------------------------------------------------
__global__ __launch_bounds__(256) void attn_score_kernel(
    const float* __restrict__ h, const float* __restrict__ w2,
    const float* __restrict__ b2, float* __restrict__ s)
{
  int t = blockIdx.x * 256 + threadIdx.x;
  if (t >= SEQ) return;
  float acc = b2[0];
  #pragma unroll 4
  for (int j = 0; j < ATTD; ++j) acc += h[(size_t)t * ATTD + j] * w2[j];
  s[t] = acc;
}

__global__ __launch_bounds__(256) void softmax_kernel(float* __restrict__ s)
{
  __shared__ float red[256];
  int tid = threadIdx.x;
  float m = -1e30f;
  for (int t = tid; t < SEQ; t += 256) m = fmaxf(m, s[t]);
  red[tid] = m; __syncthreads();
  for (int o = 128; o > 0; o >>= 1) { if (tid < o) red[tid] = fmaxf(red[tid], red[tid + o]); __syncthreads(); }
  m = red[0]; __syncthreads();
  float sum = 0.f;
  for (int t = tid; t < SEQ; t += 256) { float e = __expf(s[t] - m); s[t] = e; sum += e; }
  red[tid] = sum; __syncthreads();
  for (int o = 128; o > 0; o >>= 1) { if (tid < o) red[tid] += red[tid + o]; __syncthreads(); }
  float inv = 1.f / red[0]; __syncthreads();
  for (int t = tid; t < SEQ; t += 256) s[t] *= inv;
}

__global__ __launch_bounds__(256) void weighted_sum_kernel(
    const float* __restrict__ p, const float* __restrict__ f, float* __restrict__ out)
{
  int c = blockIdx.x * 256 + threadIdx.x;
  if (c >= DMODEL) return;
  float acc = 0.f;
  for (int t = 0; t < SEQ; ++t) acc += p[t] * f[(size_t)t * DMODEL + c];
  out[c] = acc;
}

// ---------------------------------------------------------------------------
// Host side
// ---------------------------------------------------------------------------
static void launch_gemm(hipStream_t s,
                        const float* A, int lda, const float* W, int ldw,
                        const float* bias, float* Out, int ldo,
                        int M, int N, int K, int act,
                        unsigned short* APK, unsigned short* WPK)
{
  const int kSteps = K / 32;
  const int aThreads = (M / 16) * kSteps * 32;
  const int wThreads = (N / 16) * kSteps * 32;
  pack_a_kernel<<<aThreads / 256, 256, 0, s>>>(A, lda, kSteps, APK);
  pack_w_kernel<<<wThreads / 256, 256, 0, s>>>(W, ldw, kSteps, WPK);
  dim3 grid(N / 64, M / 128);
  gemm_frag_wmma_kernel<<<grid, 256, 0, s>>>(APK, WPK, bias, Out, ldo, kSteps, act);
}

static void run_mamba(hipStream_t s, const float* x, void* const* w,
                      float* XZ, float* XIN, float* XDBL, float* DT,
                      float* YB, float* PA, float* HE, float* CY, float* F,
                      unsigned short* APK, unsigned short* WPK)
{
  const float* in_w    = (const float*)w[0];
  const float* conv_w  = (const float*)w[1];
  const float* conv_b  = (const float*)w[2];
  const float* xproj_w = (const float*)w[3];
  const float* dt_w    = (const float*)w[4];
  const float* dt_b    = (const float*)w[5];
  const float* A_log   = (const float*)w[6];
  const float* Dp      = (const float*)w[7];
  const float* out_w   = (const float*)w[8];

  // xz = x @ in_w.T  [4096, 2048]
  launch_gemm(s, x, DMODEL, in_w, DMODEL, nullptr, XZ, 2 * DINNER,
              SEQ, 2 * DINNER, DMODEL, 0, APK, WPK);
  // depthwise conv + silu -> XIN [4096, 1024]
  conv_silu_kernel<<<(SEQ * DINNER) / 256, 256, 0, s>>>(XZ, conv_w, conv_b, XIN);
  // xdbl = xin @ xproj_w.T  [4096, 64]
  launch_gemm(s, XIN, DINNER, xproj_w, DINNER, nullptr, XDBL, DTRANK + 2 * DSTATE,
              SEQ, 64, DINNER, 0, APK, WPK);
  // dt = softplus(xdbl[:, :32] @ dt_w.T + dt_b)  [4096, 1024]
  launch_gemm(s, XDBL, DTRANK + 2 * DSTATE, dt_w, DTRANK, dt_b, DT, DINNER,
              SEQ, DINNER, DTRANK, 3, APK, WPK);
  // chunked selective scan (+ D skip + silu(z) gating fused) -> YB
  scan_local_kernel<<<dim3(DINNER / 256, NCHUNK), 256, 0, s>>>(DT, XDBL, XIN, A_log, PA, HE);
  scan_carry_kernel<<<DINNER / 256, 256, 0, s>>>(PA, HE, CY);
  scan_apply_kernel<<<dim3(DINNER / 256, NCHUNK), 256, 0, s>>>(DT, XDBL, XIN, A_log, CY, Dp, XZ, YB);
  // out = (y * silu(z)) @ out_w.T  [4096, 512]
  launch_gemm(s, YB, DINNER, out_w, DINNER, nullptr, F, DMODEL,
              SEQ, DMODEL, DINNER, 0, APK, WPK);
}

extern "C" void kernel_launch(void* const* d_in, const int* in_sizes, int n_in,
                              void* d_out, int out_size, void* d_ws, size_t ws_size,
                              hipStream_t stream)
{
  (void)in_sizes; (void)n_in; (void)out_size; (void)ws_size;
  const float* x       = (const float*)d_in[0];
  const float* attn_w1 = (const float*)d_in[19];
  const float* attn_b1 = (const float*)d_in[20];
  const float* attn_w2 = (const float*)d_in[21];
  const float* attn_b2 = (const float*)d_in[22];

  float* ws = (float*)d_ws;
  size_t o = 0;
  float* XZ   = ws + o; o += (size_t)SEQ * 2 * DINNER;          // 8M f32
  float* XIN  = ws + o; o += (size_t)SEQ * DINNER;              // 4M
  float* XDBL = ws + o; o += (size_t)SEQ * (DTRANK + 2*DSTATE); // 256K
  float* DT   = ws + o; o += (size_t)SEQ * DINNER;              // 4M
  float* YB   = ws + o; o += (size_t)SEQ * DINNER;              // 4M
  float* PA   = ws + o; o += (size_t)NCHUNK * DINNER * DSTATE;  // 1M
  float* HE   = ws + o; o += (size_t)NCHUNK * DINNER * DSTATE;  // 1M
  float* CY   = ws + o; o += (size_t)NCHUNK * DINNER * DSTATE;  // 1M
  float* F1   = ws + o; o += (size_t)SEQ * DMODEL;              // 2M
  float* F2   = ws + o; o += (size_t)SEQ * DMODEL;              // 2M
  float* AH   = ws + o; o += (size_t)SEQ * ATTD;                // 512K
  float* SC   = ws + o; o += (size_t)SEQ;                       // 4K
  // packed bf16 fragment buffers (ushort), 32B aligned since o is even
  unsigned short* APK = (unsigned short*)(ws + o);              // 4M ushorts
  o += ((size_t)SEQ * DINNER) / 2;
  unsigned short* WPK = (unsigned short*)(ws + o);              // 1M ushorts
  o += ((size_t)(2 * DINNER) * DMODEL) / 2;

  // Mamba layer 1: x -> F1
  run_mamba(stream, x, d_in + 1,  XZ, XIN, XDBL, DT, YB, PA, HE, CY, F1, APK, WPK);
  // Mamba layer 2: F1 -> F2
  run_mamba(stream, F1, d_in + 10, XZ, XIN, XDBL, DT, YB, PA, HE, CY, F2, APK, WPK);

  // attention pooling over the ORIGINAL input x
  launch_gemm(stream, x, DMODEL, attn_w1, DMODEL, attn_b1, AH, ATTD,
              SEQ, ATTD, DMODEL, 2, APK, WPK);
  attn_score_kernel<<<SEQ / 256, 256, 0, stream>>>(AH, attn_w2, attn_b2, SC);
  softmax_kernel<<<1, 256, 0, stream>>>(SC);
  weighted_sum_kernel<<<(DMODEL + 255) / 256, 256, 0, stream>>>(SC, F2, (float*)d_out);
}
